// AttentionHead_49151605735548
// MI455X (gfx1250) — matrix operand order repaired and checked by reading, not compiled, not run on hardware
//
#include <hip/hip_runtime.h>

// ---------------------------------------------------------------------------
// AttentionHead for MI455X (gfx1250): bf16 WMMA flash attention.
//   B=4, S=4096, E=1024, D=64.  All matrix math on v_wmma_f32_16x16x32_bf16.
// ---------------------------------------------------------------------------

typedef __attribute__((ext_vector_type(16))) __bf16 v16bf;
typedef __attribute__((ext_vector_type(8)))  __bf16 v8bf;
typedef __attribute__((ext_vector_type(8)))  float  v8f;

#define BATCH 4
#define SEQ   4096
#define EMB   1024
#define HD    64

__device__ __forceinline__ v8f wmma_bf16(v16bf a, v16bf b, v8f c) {
    // (neg_a, A, neg_b, B, c_mod, C, reuse_a, reuse_b)
    return __builtin_amdgcn_wmma_f32_16x16x32_bf16(false, a, false, b,
                                                   (short)0, c, false, false);
}

__device__ __forceinline__ v16bf cvt16(float4 a, float4 b, float4 c, float4 d) {
    v16bf r;
    r[0]=(__bf16)a.x;  r[1]=(__bf16)a.y;  r[2]=(__bf16)a.z;  r[3]=(__bf16)a.w;
    r[4]=(__bf16)b.x;  r[5]=(__bf16)b.y;  r[6]=(__bf16)b.z;  r[7]=(__bf16)b.w;
    r[8]=(__bf16)c.x;  r[9]=(__bf16)c.y;  r[10]=(__bf16)c.z; r[11]=(__bf16)c.w;
    r[12]=(__bf16)d.x; r[13]=(__bf16)d.y; r[14]=(__bf16)d.z; r[15]=(__bf16)d.w;
    return r;
}

// ---------------------------------------------------------------------------
// Kernel 1: fused QKV projection.  grid = B*S/16, block = 128 (4 waves).
// Each wave computes the 16(M) x 16(N) tiles of Q, K and V for its column
// group with a SINGLE shared A-operand per K-step (3 WMMAs per load of A).
//   Q, K row-major bf16 [B*S][64];  V transposed bf16 [B][64][S].
// ---------------------------------------------------------------------------
__global__ __launch_bounds__(128) void qkv_proj_kernel(
    const float* __restrict__ hidden,
    const float* __restrict__ qw, const float* __restrict__ qb,
    const float* __restrict__ kw, const float* __restrict__ kb,
    const float* __restrict__ vw, const float* __restrict__ vb,
    __bf16* __restrict__ Qbf, __bf16* __restrict__ Kt,
    __bf16* __restrict__ Vbf)
{
    const int tid  = threadIdx.x;
    const int wave = tid >> 5;
    const int lane = tid & 31;
    const int half = lane >> 4;        // 0: lanes 0-15, 1: lanes 16-31
    const int lr   = lane & 15;

    const int rowbase = blockIdx.x * 16;        // flat row in [0, B*S)
    const int colbase = wave * 16;              // output column tile

    // A-operand row for this lane (M = lane & 15 for both halves)
    const float* hrow = hidden + (size_t)(rowbase + lr) * EMB;
    const int ca = half ? 8 : 0;                // K sub-chunk per half

    // B-operand: lane = N column (lr), K rows 8*half+{0..7,16..23}
    const size_t woff = (size_t)(8 * half) * HD + colbase + lr;
    const float* qwp = qw + woff;
    const float* kwp = kw + woff;
    const float* vwp = vw + woff;

    v8f cq = {}, ck = {}, cv = {};

    for (int e0 = 0; e0 < EMB; e0 += 32) {
        // ---- A: hidden 16x32 tile, bf16 A-layout (loaded ONCE, used 3x) ----
        const float* ap = hrow + e0 + ca;
        v16bf A = cvt16(*(const float4*)(ap),      *(const float4*)(ap + 4),
                        *(const float4*)(ap + 16), *(const float4*)(ap + 20));

        const size_t we = (size_t)e0 * HD;
        {   const float* wp = qwp + we;
            v16bf Bq = cvt16(make_float4(wp[0], wp[HD], wp[2*HD], wp[3*HD]), make_float4(wp[4*HD], wp[5*HD], wp[6*HD], wp[7*HD]),
                             make_float4(wp[16*HD], wp[17*HD], wp[18*HD], wp[19*HD]), make_float4(wp[20*HD], wp[21*HD], wp[22*HD], wp[23*HD]));
            cq = wmma_bf16(A, Bq, cq);
        }
        {   const float* wp = kwp + we;
            v16bf Bk = cvt16(make_float4(wp[0], wp[HD], wp[2*HD], wp[3*HD]), make_float4(wp[4*HD], wp[5*HD], wp[6*HD], wp[7*HD]),
                             make_float4(wp[16*HD], wp[17*HD], wp[18*HD], wp[19*HD]), make_float4(wp[20*HD], wp[21*HD], wp[22*HD], wp[23*HD]));
            ck = wmma_bf16(A, Bk, ck);
        }
        {   const float* wp = vwp + we;
            v16bf Bv = cvt16(make_float4(wp[0], wp[HD], wp[2*HD], wp[3*HD]), make_float4(wp[4*HD], wp[5*HD], wp[6*HD], wp[7*HD]),
                             make_float4(wp[16*HD], wp[17*HD], wp[18*HD], wp[19*HD]), make_float4(wp[20*HD], wp[21*HD], wp[22*HD], wp[23*HD]));
            cv = wmma_bf16(A, Bv, cv);
        }
    }

    // Bias: lane's column is fixed for all 8 accumulated rows.
    const int n   = colbase + lr;
    const float bnq = qb[n], bnk = kb[n], bnv = vb[n];

    // Store C (VGPR i -> row = rowbase + i + 8*half, col = n)
    #pragma unroll
    for (int i = 0; i < 8; ++i) {
        const int row  = rowbase + i + 8 * half;      // flat b*S + s
        const int bidx = row >> 12;                   // / SEQ
        const int s    = row & (SEQ - 1);
        Qbf[(size_t)row * HD + n]             = (__bf16)(cq[i] + bnq);
        Kt[(size_t)row * HD + n]              = (__bf16)(ck[i] + bnk);
        Vbf[((size_t)bidx * HD + n) * SEQ + s] = (__bf16)(cv[i] + bnv);
    }
}

// ---------------------------------------------------------------------------
// Kernel 2: flash attention.  grid = B*S/(16*4), block = 128 (4 waves).
// Each wave owns a 16-query tile, streams keys 64 at a time:
//   8 WMMA for S = Q K^T, online softmax (reductions amortized over 64 keys),
//   LDS transpose of P, 8 WMMA for P V.
// ---------------------------------------------------------------------------
__global__ __launch_bounds__(128) void flash_attn_kernel(
    const __bf16* __restrict__ Qbf,   // [B*S][64] row-major
    const __bf16* __restrict__ Kt,    // [B*S][64] row-major
    const __bf16* __restrict__ Vbf,   // [B][64][S]
    float* __restrict__ out)          // [B*S][64]
{
    __shared__ float lds_p[4][16 * 64];   // per-wave 16x64 P staging (16 KB)

    const int tid  = threadIdx.x;
    const int wave = tid >> 5;
    const int lane = tid & 31;
    const int half = lane >> 4;
    const int lr   = lane & 15;

    const int tile  = blockIdx.x * 4 + wave;   // 0 .. B*S/16 - 1
    const int b     = tile >> 8;               // S/16 = 256 tiles per batch
    const int qrow0 = (tile & 255) * 16;

    // ---- Load Q A-operands (16x64 split into two 16x32 bf16 operands) ----
    const __bf16* qp = Qbf + (size_t)(b * SEQ + qrow0 + lr) * HD;
    const int ca = half ? 8 : 0;
    v8bf q00 = *(const v8bf*)(qp + ca);
    v8bf q01 = *(const v8bf*)(qp + 16 + ca);
    v8bf q10 = *(const v8bf*)(qp + 32 + ca);
    v8bf q11 = *(const v8bf*)(qp + 48 + ca);
    v16bf qlo = __builtin_shufflevector(q00, q01, 0,1,2,3,4,5,6,7,8,9,10,11,12,13,14,15);
    v16bf qhi = __builtin_shufflevector(q10, q11, 0,1,2,3,4,5,6,7,8,9,10,11,12,13,14,15);

    // K B-operand lane pointers: lane's key = lr, d = ca+{0..7,16..23} (+32 for hi chunk)
    const __bf16* kt_lo = Kt + (size_t)(b * SEQ + lr) * HD + ca;
    const __bf16* kt_hi = kt_lo + 32;

    // Online softmax state, per lane: 8 rows (row = qrow0 + i + 8*half)
    float m[8], lsum[8];
    v8f acc[4];
    #pragma unroll
    for (int i = 0; i < 8; ++i) { m[i] = -3.0e38f; lsum[i] = 0.0f; }
    #pragma unroll
    for (int j = 0; j < 4; ++j) acc[j] = (v8f){};

    // scale = 1/sqrt(64) folded with log2(e) so we can use exp2
    const float SC = 0.125f * 1.44269504088896340736f;

    float* lp = &lds_p[wave][0];

    for (int kb = 0; kb < SEQ; kb += 64) {
        // Prefetch next iteration's K operand rows (global_prefetch_b8)
        if (kb + 64 < SEQ) {
            __builtin_prefetch(kt_lo + (size_t)(kb + 64) * HD, 0, 3);
            __builtin_prefetch(kt_hi + (size_t)(kb + 64) * HD, 0, 3);
        }

        // ---- scores: 16 queries x 64 keys (8 WMMA) ----
        v8f s[4];
        #pragma unroll
        for (int t = 0; t < 4; ++t) {
            const __bf16* kr = kt_lo + (size_t)(kb + 16 * t) * HD;
            v16bf klo = __builtin_shufflevector(*(const v8bf*)(kr), *(const v8bf*)(kr + 16), 0,1,2,3,4,5,6,7,8,9,10,11,12,13,14,15);
            v16bf khi = __builtin_shufflevector(*(const v8bf*)(kr + 32), *(const v8bf*)(kr + 48), 0,1,2,3,4,5,6,7,8,9,10,11,12,13,14,15);
            v8f st = {};
            st = wmma_bf16(qlo, klo, st);
            st = wmma_bf16(qhi, khi, st);
            s[t] = st;
        }

        // ---- online softmax (row reductions across 16-lane halves) ----
        float alpha[8];
        #pragma unroll
        for (int i = 0; i < 8; ++i) {
            float a0 = s[0][i] * SC;
            float a1 = s[1][i] * SC;
            float a2 = s[2][i] * SC;
            float a3 = s[3][i] * SC;
            float mx = fmaxf(fmaxf(a0, a1), fmaxf(a2, a3));
            mx = fmaxf(mx, __shfl_xor(mx, 1, 32));
            mx = fmaxf(mx, __shfl_xor(mx, 2, 32));
            mx = fmaxf(mx, __shfl_xor(mx, 4, 32));
            mx = fmaxf(mx, __shfl_xor(mx, 8, 32));
            const float mnew = fmaxf(m[i], mx);
            alpha[i] = exp2f(m[i] - mnew);
            const float p0 = exp2f(a0 - mnew);
            const float p1 = exp2f(a1 - mnew);
            const float p2 = exp2f(a2 - mnew);
            const float p3 = exp2f(a3 - mnew);
            float rs = (p0 + p1) + (p2 + p3);
            rs += __shfl_xor(rs, 1, 32);
            rs += __shfl_xor(rs, 2, 32);
            rs += __shfl_xor(rs, 4, 32);
            rs += __shfl_xor(rs, 8, 32);
            lsum[i] = lsum[i] * alpha[i] + rs;
            m[i] = mnew;
            s[0][i] = p0;
            s[1][i] = p1;
            s[2][i] = p2;
            s[3][i] = p3;
        }
        #pragma unroll
        for (int j = 0; j < 4; ++j)
            #pragma unroll
            for (int i = 0; i < 8; ++i) acc[j][i] *= alpha[i];

        // ---- transpose P from C-layout to A-layout via wave-private LDS ----
        #pragma unroll
        for (int i = 0; i < 8; ++i) {
            const int row = i + 8 * half;
            lp[row * 64 + lr]      = s[0][i];
            lp[row * 64 + 16 + lr] = s[1][i];
            lp[row * 64 + 32 + lr] = s[2][i];
            lp[row * 64 + 48 + lr] = s[3][i];
        }
        asm volatile("s_wait_dscnt 0x0" ::: "memory");

        const float* pr = lp + lr * 64;      // A row = lane & 15
        v16bf pA0 = cvt16(*(const float4*)(pr + ca),      *(const float4*)(pr + ca + 4),
                          *(const float4*)(pr + ca + 16), *(const float4*)(pr + ca + 20));
        v16bf pA1 = cvt16(*(const float4*)(pr + ca + 32), *(const float4*)(pr + ca + 36),
                          *(const float4*)(pr + ca + 48), *(const float4*)(pr + ca + 52));

        // ---- P @ V : B-operand lane d = 16*j + lr, keys kb(+32) + ca+{0..7,16..23} ----
        const __bf16* vp0 = Vbf + ((size_t)b * HD + lr) * SEQ + kb + ca;
        const __bf16* vp1 = vp0 + 32;
        if (kb + 64 < SEQ) __builtin_prefetch(vp0 + 64, 0, 3);
        #pragma unroll
        for (int j = 0; j < 4; ++j) {
            const __bf16* vr = vp0 + (size_t)16 * j * SEQ;
            v16bf Bv0 = __builtin_shufflevector(*(const v8bf*)(vr), *(const v8bf*)(vr + 16), 0,1,2,3,4,5,6,7,8,9,10,11,12,13,14,15);
            acc[j] = wmma_bf16(pA0, Bv0, acc[j]);
        }
        #pragma unroll
        for (int j = 0; j < 4; ++j) {
            const __bf16* vr = vp1 + (size_t)16 * j * SEQ;
            v16bf Bv1 = __builtin_shufflevector(*(const v8bf*)(vr), *(const v8bf*)(vr + 16), 0,1,2,3,4,5,6,7,8,9,10,11,12,13,14,15);
            acc[j] = wmma_bf16(pA1, Bv1, acc[j]);
        }
    }

    // ---- epilogue: normalize and store f32 ----
    float inv[8];
    #pragma unroll
    for (int i = 0; i < 8; ++i) inv[i] = 1.0f / lsum[i];
    #pragma unroll
    for (int j = 0; j < 4; ++j)
        #pragma unroll
        for (int i = 0; i < 8; ++i) {
            const int row = qrow0 + i + 8 * half;
            out[(size_t)(b * SEQ + row) * HD + 16 * j + lr] = acc[j][i] * inv[i];
        }
}

// ---------------------------------------------------------------------------
extern "C" void kernel_launch(void* const* d_in, const int* in_sizes, int n_in,
                              void* d_out, int out_size, void* d_ws, size_t ws_size,
                              hipStream_t stream) {
    const float* hidden = (const float*)d_in[0];
    const float* qw = (const float*)d_in[1];
    const float* qb = (const float*)d_in[2];
    const float* kw = (const float*)d_in[3];
    const float* kb = (const float*)d_in[4];
    const float* vw = (const float*)d_in[5];
    const float* vb = (const float*)d_in[6];
    float* out = (float*)d_out;

    __bf16* Qbf = (__bf16*)d_ws;                              // 2 MB
    __bf16* Kt  = Qbf + (size_t)BATCH * SEQ * HD;             // 2 MB
    __bf16* Vbf = Kt  + (size_t)BATCH * HD * SEQ;             // 2 MB

    qkv_proj_kernel<<<BATCH * SEQ / 16, 128, 0, stream>>>(hidden, qw, qb, kw, kb,
                                                          vw, vb, Qbf, Kt, Vbf);

    flash_attn_kernel<<<BATCH * SEQ / (16 * 4), 128, 0, stream>>>(Qbf, Kt, Vbf, out);
}
